// BatchMuSc_54314156425484
// MI455X (gfx1250) — matrix operand order repaired and checked by reading, not compiled, not run on hardware
//
#include <hip/hip_runtime.h>
#include <hip/hip_bf16.h>
#include <math.h>

// ---------------------------------------------------------------------------
// Problem constants (from reference): N=96 images, L=256 patches, C=128 ch,
// DC=768 cls dim.  k_max = int(95*0.3) = 28, k_min = 0.
// ---------------------------------------------------------------------------
#define NIMG 96
#define LPAT 256
#define CCH  128
#define DCLS 768
#define NPAT (NIMG * LPAT)   // 24576
#define KSEL 28

typedef __attribute__((ext_vector_type(16))) __bf16 v16bf;
typedef __attribute__((ext_vector_type(8)))  float  v8f;

union FragCvt { uint4 q[2]; v16bf v; };

__device__ __forceinline__ v8f zero8() {
  v8f z = {0.f, 0.f, 0.f, 0.f, 0.f, 0.f, 0.f, 0.f};
  return z;
}

// A-fragment (16x32 bf16), LDS stride 72 shorts (padded).
// lane<16: elems 0..7 = K+0..7, 8..15 = K+16..23 (row = rowBase+lane)
// lane>=16: elems 0..7 = K+8..15, 8..15 = K+24..31 (row = rowBase+lane-16)
__device__ __forceinline__ v16bf load_frag_a(const unsigned short* sm,
                                             int rowBase, int kb, int lane) {
  const int r = rowBase + (lane & 15);
  const int c = kb + ((lane >> 4) << 3);          // +8 for upper half
  const unsigned short* ptr = sm + r * 72 + c;
  FragCvt f;
  f.q[0] = *(const uint4*)(ptr);                  // K + [0..7]
  f.q[1] = *(const uint4*)(ptr + 16);             // K + 16 + [0..7]
  return f.v;
}

// B-fragment (32x16 bf16) for B = Zj^T: lane holds one patch row of Zj.
// lane<16: elems 0..7 = channels K+0..7, 8..15 = K+16..23 of patch colBase+lane
// lane>=16: elems 0..7 = channels K+8..15, 8..15 = K+24..31 of patch colBase+lane-16
__device__ __forceinline__ v16bf load_frag_b(const unsigned short* sm,
                                             int colBase, int kb, int lane) {
  const int r = colBase + (lane & 15);
  const int c = kb + ((lane >> 4) << 3);          // +8 for upper half
  const unsigned short* ptr = sm + r * 72 + c;
  FragCvt f;
  f.q[0] = *(const uint4*)(ptr);
  f.q[1] = *(const uint4*)(ptr + 16);
  return f.v;
}

__device__ __forceinline__ unsigned short f32_to_bf16_rne(float x) {
  unsigned u = __float_as_uint(x);
  unsigned r = u + 0x7FFFu + ((u >> 16) & 1u);
  return (unsigned short)(r >> 16);
}

// ---------------------------------------------------------------------------
// Kernel 1: split fp32 -> bf16 hi/lo, per-patch squared norm.
// grid = NPAT blocks, 128 threads (one channel each).
// ---------------------------------------------------------------------------
__global__ void musc_prep_kernel(const float* __restrict__ Z,
                                 unsigned short* __restrict__ Zhi,
                                 unsigned short* __restrict__ Zlo,
                                 float* __restrict__ sq) {
  const int p = blockIdx.x;
  const int c = threadIdx.x;
  const float z = Z[(size_t)p * CCH + c];
  const unsigned short h = f32_to_bf16_rne(z);
  const float hf = __uint_as_float(((unsigned)h) << 16);
  const unsigned short lo = f32_to_bf16_rne(z - hf);
  Zhi[(size_t)p * CCH + c] = h;
  Zlo[(size_t)p * CCH + c] = lo;

  __shared__ float red[128];
  red[c] = z * z;
  __syncthreads();
  for (int s = 64; s > 0; s >>= 1) {
    if (c < s) red[c] += red[c + s];
    __syncthreads();
  }
  if (c == 0) sq[p] = red[0];
}

// ---------------------------------------------------------------------------
// Kernel 2: init minD2 to +inf bit pattern.
// ---------------------------------------------------------------------------
__global__ void musc_init_kernel(unsigned* __restrict__ minD2, int n) {
  int t = blockIdx.x * blockDim.x + threadIdx.x;
  if (t < n) minD2[t] = 0x7F800000u;   // +inf
}

// ---------------------------------------------------------------------------
// Kernel 3: the big one.  Per block: image-pair (i, j), 64 rows (patches of i)
// x 128 cols (patches of j), K = 128 staged in LDS as two 64-wide stages.
// Split-bf16 GEMM (hi*hi + hi*lo + lo*hi) via v_wmma_f32_16x16x32_bf16,
// then per-row min over columns -> atomicMin(uint bits) into minD2[p][j].
// grid = 96*96*4*2 = 73728 blocks, 256 threads (8 waves, wave32).
// ---------------------------------------------------------------------------
__global__ __launch_bounds__(256)
void musc_gemm_min_kernel(const unsigned short* __restrict__ Zhi,
                          const unsigned short* __restrict__ Zlo,
                          const float* __restrict__ sq,
                          float* __restrict__ minD2) {
  __shared__ __attribute__((aligned(16))) unsigned short sAhi[64 * 72];
  __shared__ __attribute__((aligned(16))) unsigned short sAlo[64 * 72];
  __shared__ __attribute__((aligned(16))) unsigned short sBhi[128 * 72];
  __shared__ __attribute__((aligned(16))) unsigned short sBlo[128 * 72];

  const int b  = blockIdx.x;
  const int nb = b & 1;            // column half of image j
  const int mb = (b >> 1) & 3;     // row quarter of image i
  const int j  = (b >> 3) % NIMG;
  const int i  = (b >> 3) / NIMG;
  if (i == j) return;              // self image excluded -> minD2 stays +inf

  const int tid  = threadIdx.x;
  const int lane = tid & 31;
  const int wave = tid >> 5;
  const int wm   = wave >> 2;      // 0..1 : 32-row band
  const int wn   = wave & 3;       // 0..3 : 32-col band

  const int p0 = i * LPAT + mb * 64;
  const int q0 = j * LPAT + nb * 128;

  v8f acc[2][2];
#pragma unroll
  for (int a = 0; a < 2; ++a)
#pragma unroll
    for (int c = 0; c < 2; ++c) acc[a][c] = zero8();

#pragma unroll
  for (int s = 0; s < 2; ++s) {      // K stages of 64 channels
    const int kk = s * 64;
    // Stage A tile: 64 rows x 64 ch, 16B chunks
#pragma unroll
    for (int t = 0; t < 2; ++t) {
      const int ch = tid + t * 256;          // 0..511
      const int r  = ch >> 3;
      const int c  = (ch & 7) << 3;
      const size_t g = (size_t)(p0 + r) * CCH + kk + c;
      *(uint4*)(sAhi + r * 72 + c) = *(const uint4*)(Zhi + g);
      *(uint4*)(sAlo + r * 72 + c) = *(const uint4*)(Zlo + g);
    }
    // Stage B tile: 128 rows x 64 ch
#pragma unroll
    for (int t = 0; t < 4; ++t) {
      const int ch = tid + t * 256;          // 0..1023
      const int r  = ch >> 3;
      const int c  = (ch & 7) << 3;
      const size_t g = (size_t)(q0 + r) * CCH + kk + c;
      *(uint4*)(sBhi + r * 72 + c) = *(const uint4*)(Zhi + g);
      *(uint4*)(sBlo + r * 72 + c) = *(const uint4*)(Zlo + g);
    }
    __syncthreads();

#pragma unroll
    for (int k32 = 0; k32 < 64; k32 += 32) {
      v16bf ah[2], al[2], bh[2], bl[2];
#pragma unroll
      for (int ms = 0; ms < 2; ++ms) {
        ah[ms] = load_frag_a(sAhi, wm * 32 + ms * 16, k32, lane);
        al[ms] = load_frag_a(sAlo, wm * 32 + ms * 16, k32, lane);
      }
#pragma unroll
      for (int ns = 0; ns < 2; ++ns) {
        bh[ns] = load_frag_b(sBhi, wn * 32 + ns * 16, k32, lane);
        bl[ns] = load_frag_b(sBlo, wn * 32 + ns * 16, k32, lane);
      }
#pragma unroll
      for (int ms = 0; ms < 2; ++ms)
#pragma unroll
        for (int ns = 0; ns < 2; ++ns) {
          acc[ms][ns] = __builtin_amdgcn_wmma_f32_16x16x32_bf16(
              false, ah[ms], false, bh[ns], (short)0, acc[ms][ns], false, false);
          acc[ms][ns] = __builtin_amdgcn_wmma_f32_16x16x32_bf16(
              false, ah[ms], false, bl[ns], (short)0, acc[ms][ns], false, false);
          acc[ms][ns] = __builtin_amdgcn_wmma_f32_16x16x32_bf16(
              false, al[ms], false, bh[ns], (short)0, acc[ms][ns], false, false);
        }
    }
    __syncthreads();
  }

  // Epilogue: d2 = |zi|^2 + |zj|^2 - 2*dot; row-min over this block's columns.
  // C/D layout: element v of lane l -> row = base_m + v + 8*(l>=16),
  //                                    col = base_n + (l&15).
  const int half = lane >> 4;
  const int colL = lane & 15;
  const float sq_q0 = sq[q0 + wn * 32 + 0 * 16 + colL];
  const float sq_q1 = sq[q0 + wn * 32 + 1 * 16 + colL];
#pragma unroll
  for (int ms = 0; ms < 2; ++ms) {
    const int pBase = p0 + wm * 32 + ms * 16 + half * 8;
#pragma unroll
    for (int v = 0; v < 8; ++v) {
      const int p = pBase + v;
      const float sp = sq[p];
      const float d0 = fmaxf(sp + sq_q0 - 2.0f * acc[ms][0][v], 0.0f);
      const float d1 = fmaxf(sp + sq_q1 - 2.0f * acc[ms][1][v], 0.0f);
      float m = fminf(d0, d1);
      // min across the 16 lanes of this half (same row, 16 distinct columns)
      m = fminf(m, __shfl_xor(m, 1, 32));
      m = fminf(m, __shfl_xor(m, 2, 32));
      m = fminf(m, __shfl_xor(m, 4, 32));
      m = fminf(m, __shfl_xor(m, 8, 32));
      if (colL == 0) {
        // non-negative floats: uint order == float order
        atomicMin((unsigned*)(minD2 + (size_t)p * NIMG + j), __float_as_uint(m));
      }
    }
  }
}

// ---------------------------------------------------------------------------
// Kernel 4: per patch: dist = sqrt(minD2 + 1e-12); mean of 28 smallest of 96
// (self image is +inf so auto-excluded); block-max over 256 patches -> image.
// grid = 96, block = 256.
// ---------------------------------------------------------------------------
__global__ void musc_scores_kernel(const float* __restrict__ minD2,
                                   float* __restrict__ imgScores) {
  const int i = blockIdx.x;
  const int l = threadIdx.x;
  const size_t p = (size_t)i * LPAT + l;
  float arr[NIMG];
  for (int jj = 0; jj < NIMG; ++jj)
    arr[jj] = sqrtf(fmaxf(minD2[p * NIMG + jj], 0.0f) + 1e-12f);

  float sum = 0.f;
  for (int t = 0; t < KSEL; ++t) {
    float mn = arr[0];
    int mi = 0;
    for (int jj = 1; jj < NIMG; ++jj)
      if (arr[jj] < mn) { mn = arr[jj]; mi = jj; }
    sum += mn;
    arr[mi] = __uint_as_float(0x7F800000u);
  }
  const float score = sum * (1.0f / (float)KSEL);

  __shared__ float red[256];
  red[l] = score;
  __syncthreads();
  for (int s = 128; s > 0; s >>= 1) {
    if (l < s) red[l] = fmaxf(red[l], red[l + s]);
    __syncthreads();
  }
  if (l == 0) imgScores[i] = red[0];
}

// ---------------------------------------------------------------------------
// Kernel 5: RsCIN finale. Single block. Min-max normalize, W = cls@cls^T,
// per-row top-k (k=1,2,3) threshold mask, row-normalize, P@s, mean.
// block = 128 threads (96 active rows).
// ---------------------------------------------------------------------------
__global__ void musc_final_kernel(const float* __restrict__ cls,
                                  const float* __restrict__ imgScores,
                                  float* __restrict__ out) {
  __shared__ float s[NIMG];
  __shared__ float Wrow[NIMG][NIMG + 1];
  __shared__ float mnmx[2];
  const int t = threadIdx.x;

  if (t == 0) {
    float mn = imgScores[0], mx = imgScores[0];
    for (int a = 1; a < NIMG; ++a) {
      const float v = imgScores[a];
      mn = fminf(mn, v);
      mx = fmaxf(mx, v);
    }
    mnmx[0] = mn;
    mnmx[1] = mx;
  }
  __syncthreads();
  if (t < NIMG) s[t] = (imgScores[t] - mnmx[0]) / (mnmx[1] - mnmx[0]);
  __syncthreads();

  if (t < NIMG) {
    const float* ct = cls + (size_t)t * DCLS;
    for (int jj = 0; jj < NIMG; ++jj) {
      const float* cj = cls + (size_t)jj * DCLS;
      float d = 0.f;
      for (int c = 0; c < DCLS; ++c) d = fmaf(ct[c], cj[c], d);
      Wrow[t][jj] = d;
    }
    float m1 = -INFINITY, m2 = -INFINITY, m3 = -INFINITY;
    for (int jj = 0; jj < NIMG; ++jj) {
      const float w = Wrow[t][jj];
      if (w > m1)      { m3 = m2; m2 = m1; m1 = w; }
      else if (w > m2) { m3 = m2; m2 = w; }
      else if (w > m3) { m3 = w; }
    }
    const float th[3] = {m1, m2, m3};
    float acc = 0.f;
    for (int k = 0; k < 3; ++k) {
      float sw = 0.f, dp = 0.f;
      for (int jj = 0; jj < NIMG; ++jj) {
        const float w = Wrow[t][jj];
        if (w >= th[k]) { sw += w; dp += w * s[jj]; }
      }
      acc += dp / sw;
    }
    out[t] = acc * (1.0f / 3.0f);
  }
}

// ---------------------------------------------------------------------------
// Host-side launcher
// ---------------------------------------------------------------------------
extern "C" void kernel_launch(void* const* d_in, const int* in_sizes, int n_in,
                              void* d_out, int out_size, void* d_ws, size_t ws_size,
                              hipStream_t stream) {
  (void)in_sizes; (void)n_in; (void)out_size; (void)ws_size;
  const float* Z   = (const float*)d_in[0];   // [96,256,128] fp32
  const float* cls = (const float*)d_in[1];   // [96,768] fp32
  float* out = (float*)d_out;                 // [96] fp32

  char* ws = (char*)d_ws;
  // Workspace layout (offsets 256B-aligned), total ~22.2 MB:
  unsigned short* Zhi = (unsigned short*)(ws);                       // 6291456 B
  unsigned short* Zlo = (unsigned short*)(ws + 6291456);             // 6291456 B
  float* sq           = (float*)(ws + 12582912);                     //   98304 B
  float* minD2        = (float*)(ws + 12681216);                     // 9437184 B
  float* imgScores    = (float*)(ws + 22118400);                     //     384 B

  musc_prep_kernel<<<NPAT, 128, 0, stream>>>(Z, Zhi, Zlo, sq);
  musc_init_kernel<<<(NPAT * NIMG + 255) / 256, 256, 0, stream>>>(
      (unsigned*)minD2, NPAT * NIMG);
  musc_gemm_min_kernel<<<NIMG * NIMG * 8, 256, 0, stream>>>(Zhi, Zlo, sq, minD2);
  musc_scores_kernel<<<NIMG, 256, 0, stream>>>(minD2, imgScores);
  musc_final_kernel<<<1, 128, 0, stream>>>(cls, imgScores, out);
}
